// XEyTransformerLayer_76416058130901
// MI455X (gfx1250) — matrix operand order repaired and checked by reading, not compiled, not run on hardware
//
#include <hip/hip_runtime.h>
#include <hip/hip_bf16.h>
#include <math.h>

// ---------------- types / WMMA helpers ----------------
typedef __bf16 bf16;
typedef bf16  bf16x16 __attribute__((ext_vector_type(16)));
typedef float f32x8   __attribute__((ext_vector_type(8)));
typedef unsigned int u32x4 __attribute__((ext_vector_type(4)));
typedef int i32x4 __attribute__((ext_vector_type(4)));
typedef int i32x8 __attribute__((ext_vector_type(8)));

__device__ __forceinline__ f32x8 wmma_bf16(bf16x16 a, bf16x16 b, f32x8 c) {
  return __builtin_amdgcn_wmma_f32_16x16x32_bf16(false, a, false, b, (short)0, c, false, false);
}
// A-fragment K index map (16-bit A 16x32, ISA 7.12.2)
__device__ __forceinline__ int a_kmap(int j, int h) { return (j < 8) ? (h * 8 + j) : (8 + h * 8 + j); }

__device__ __forceinline__ bf16x16 gather_a_f32(const float* A, int lda, int lane) {
  int r = lane & 15, h = lane >> 4; bf16x16 o;
#pragma unroll
  for (int j = 0; j < 16; ++j) o[j] = (bf16)A[r * lda + a_kmap(j, h)];
  return o;
}
__device__ __forceinline__ bf16x16 gather_a_bf16(const bf16* A, int lda, int lane) {
  int r = lane & 15, h = lane >> 4; bf16x16 o;
#pragma unroll
  for (int j = 0; j < 16; ++j) o[j] = A[r * lda + a_kmap(j, h)];
  return o;
}
// B fragment: 32x16 row-major (K x N); lane = column, VGPR j = K
__device__ __forceinline__ bf16x16 gather_b_bf16(const bf16* B, int ldb, int lane) {
  int c = lane & 15, h = lane >> 4; bf16x16 o;
#pragma unroll
  for (int j = 0; j < 16; ++j) o[j] = B[a_kmap(j, h) * ldb + c];
  return o;
}
__device__ __forceinline__ float sigmoidf_(float x) { return 1.f / (1.f + __expf(-x)); }
__device__ __forceinline__ float siluf_(float x) { return x / (1.f + __expf(-x)); }
__device__ __forceinline__ float geluf_(float x) { return 0.5f * x * (1.f + erff(x * 0.70710678118f)); }

// ---------------- fp32 -> bf16 weight convert ----------------
__global__ void cvt_kernel(const float* __restrict__ s, bf16* __restrict__ d, int n) {
  int i = blockIdx.x * 256 + threadIdx.x;
  if (i < n) d[i] = (bf16)s[i];
}

// ---------------- generic WMMA GEMM: C = act(A@B + bias) * rowmask ----------------
// one wave per 16x16 tile; K multiple of 32 and <= 256; A fp32 row-major,
// B bf16 row-major (in x out). The K x 16 B-panel is brought into LDS by the
// Tensor Data Mover (one 2D-tile DMA descriptor, TENSORcnt-tracked) when the
// toolchain exposes the builtin; otherwise a manual wave copy is used.
__global__ void gemm_kernel(const float* __restrict__ A, int lda,
                            const bf16* __restrict__ B, int ldb,
                            const float* __restrict__ bias,
                            const float* __restrict__ rowmask,
                            float* __restrict__ C, int ldc, int K, int act) {
  __shared__ bf16 smB[256 * 16];  // K<=256 panel, 8KB
  int lane = threadIdx.x;
  int n0 = blockIdx.x * 16, m0 = blockIdx.y * 16;

#if defined(__AMDGCN__) && __has_builtin(__builtin_amdgcn_tensor_load_to_lds)
  {
    // D# group0: count=1 | lds_addr | global_addr(57b) | type=2  (ISA 8.3)
    unsigned lds_addr = (unsigned)(size_t)(&smB[0]);
    unsigned long long ga = (unsigned long long)(size_t)(B + (size_t)n0);
    u32x4 g0;
    g0[0] = 1u;                                              // count=1 (user D#)
    g0[1] = lds_addr;                                        // LDS dest
    g0[2] = (unsigned)(ga & 0xffffffffu);                    // global addr lo
    g0[3] = (unsigned)((ga >> 32) & 0x01ffffffu) | (2u << 30); // addr hi | type=2
    // D# group1: data_size=2B, tensor_dim0=16, tensor_dim1=K,
    // tile_dim0=16, tile_dim1=K, dim0_stride=ldb (elements)   (ISA 8.4)
    i32x8 g1;
    g1[0] = (int)(1u << 16);                                 // data_size=1 -> 2 bytes
    g1[1] = (int)(16u << 16);                                // tensor_dim0 lo16
    g1[2] = (int)(((unsigned)K & 0xffffu) << 16);            // dim0 hi | tensor_dim1 lo
    g1[3] = (int)(16u << 16);                                // dim1 hi | tile_dim0=16
    g1[4] = (int)((unsigned)K & 0xffffu);                    // tile_dim1=K | tile_dim2=0
    g1[5] = (int)(unsigned)ldb;                              // tensor_dim0_stride lo32
    g1[6] = 0;                                               // stride hi | dim1_stride lo
    g1[7] = 0;
    i32x4 z4 = {0, 0, 0, 0};
#if __clang_major__ >= 23
    i32x8 z8 = {0, 0, 0, 0, 0, 0, 0, 0};
    __builtin_amdgcn_tensor_load_to_lds(g0, g1, z4, z4, z8, 0);
#else
    __builtin_amdgcn_tensor_load_to_lds(g0, g1, z4, z4, 0);
#endif
    __builtin_amdgcn_s_wait_tensorcnt((short)0);
  }
#else
  for (int i = lane; i < K * 16; i += 32) {
    int r = i >> 4, c = i & 15;
    smB[i] = B[(size_t)r * ldb + n0 + c];
  }
  __syncthreads();
#endif

  f32x8 acc = {0.f, 0.f, 0.f, 0.f, 0.f, 0.f, 0.f, 0.f};
  for (int k0 = 0; k0 < K; k0 += 32) {
    bf16x16 af = gather_a_f32(A + (size_t)m0 * lda + k0, lda, lane);
    bf16x16 bfr = gather_b_bf16(smB + k0 * 16, 16, lane);
    acc = wmma_bf16(af, bfr, acc);
  }
  int col = n0 + (lane & 15);
  int rb = m0 + ((lane >> 4) << 3);
  float bv = bias ? bias[col] : 0.f;
#pragma unroll
  for (int j = 0; j < 8; ++j) {
    int row = rb + j;
    float v = acc[j] + bv;
    if (act == 1) v = geluf_(v);
    else if (act == 2) v = siluf_(v);
    else if (act == 3) v = fmaxf(v, 0.f);
    if (rowmask) v *= rowmask[row];
    C[(size_t)row * ldc + col] = v;
  }
}

// ---------------- dynamic per-token weights (dw path) ----------------
// dwraw: (bs*n,160); qkw: (4,32,32); outputs W1n/W2: (tok,4,2,8), DD: (tok,4,8)
__global__ void dw_kernel(const float* __restrict__ dwraw, const float* __restrict__ qkw,
                          float* __restrict__ W1n, float* __restrict__ W2, float* __restrict__ DD) {
  __shared__ float dwh[128];
  __shared__ float dws[128];
  __shared__ float rpart[64];
  __shared__ float rsum[8];
  int tok = blockIdx.x, t = threadIdx.x;
  float r = dwraw[(size_t)tok * 160 + t];
  dwh[t] = geluf_(r);
  if (t < 32) DD[(size_t)tok * 32 + t] = tanhf(dwraw[(size_t)tok * 160 + 128 + t]);
  __syncthreads();
  int c = t >> 5, d = t & 31;
  float s = 0.f;
  for (int kk = 0; kk < 32; ++kk) s += dwh[c * 32 + kk] * qkw[(c * 32 + kk) * 32 + d];
  dws[t] = s;
  __syncthreads();
  if (t < 64) {
    int cc = t >> 4, ii = (t >> 3) & 1, hh = t & 7;
    float v = dws[cc * 32 + ii * 8 + hh];
    rpart[t] = v * v;
  }
  __syncthreads();
  if (t < 8) {
    float m = 0.f;
    for (int u = 0; u < 8; ++u) m += rpart[t * 8 + u];
    rsum[t] = rsqrtf(m * 0.125f + 1e-6f);
  }
  __syncthreads();
  if (t < 64) {
    int cc = t >> 4, ii = (t >> 3) & 1, hh = t & 7;
    W1n[(size_t)tok * 64 + t] = dws[cc * 32 + ii * 8 + hh] * rsum[t >> 3];
    W2[(size_t)tok * 64 + t] = dws[cc * 32 + (2 + ii) * 8 + hh];
  }
}

// ---------------- gates from y ----------------
__global__ void gate_kernel(const float* __restrict__ y,
                            const float* __restrict__ ngw, const float* __restrict__ ngb,
                            const float* __restrict__ egw, const float* __restrict__ egb,
                            float* __restrict__ gateN, float* __restrict__ gateE) {
  int b = blockIdx.x, c = threadIdx.x;
  float an = ngb[c], ae = egb[c];
  for (int j = 0; j < 128; ++j) {
    float yv = y[b * 128 + j];
    an += yv * ngw[j * 256 + c];
    ae += yv * egw[j * 256 + c];
  }
  gateN[b * 256 + c] = sigmoidf_(an);
  gateE[b * 256 + c] = sigmoidf_(ae) + 1.f;
}

// ---------------- monster: per-(b,k) attention + edge update ----------------
__global__ void __launch_bounds__(256) edge_attn_kernel(
    const float* __restrict__ QKV, const float* __restrict__ W1n, const float* __restrict__ W2,
    const float* __restrict__ DD, const float* __restrict__ gateN, const float* __restrict__ gateE,
    const bf16* __restrict__ eoutw, const float* __restrict__ eoutb,
    const float* __restrict__ Ein, const float* __restrict__ mask,
    float* __restrict__ outE, float* __restrict__ newXpre) {
  extern __shared__ char smraw[];
  float* l_log = (float*)smraw;        // 128*256 logits -> probs
  float* l_kr = l_log + 128 * 256;     // 256 K-row
  float* l_mq = l_kr + 256;            // 128 q-mask
  float* l_ge = l_mq + 128;            // 256 edge gate (+1)
  float* l_wv = l_ge + 256;            // 256 weighted V accum
  float* l_red = l_wv + 256;           // 256 reduction
  float* l_wk = l_red + 256;           // 96: pre{w1k,w2k,ddk} post{...}
  int b = blockIdx.x >> 7, k = blockIdx.x & 127;
  int tid = threadIdx.x;
  int base_k = b * 128 + k;
  float mk = mask[base_k];
  l_kr[tid] = QKV[(size_t)base_k * 768 + 256 + tid];
  l_ge[tid] = gateE[b * 256 + tid];
  l_wv[tid] = 0.f;
  if (tid < 128) l_mq[tid] = mask[b * 128 + tid];
  if (tid < 16) {
    l_wk[tid]      = W1n[((size_t)base_k * 4 + 1) * 16 + tid];
    l_wk[16 + tid] = W2[((size_t)base_k * 4 + 1) * 16 + tid];
    l_wk[40 + tid] = W1n[((size_t)base_k * 4 + 3) * 16 + tid];
    l_wk[56 + tid] = W2[((size_t)base_k * 4 + 3) * 16 + tid];
  }
  if (tid < 8) {
    l_wk[32 + tid] = DD[((size_t)base_k * 4 + 1) * 8 + tid];
    l_wk[72 + tid] = DD[((size_t)base_k * 4 + 3) * 8 + tid];
  }
  __syncthreads();

  // Phase 1: logits = cross_head_proj(Q*K/sqrt(DF), pre)
  {
    int q = tid >> 1, f0 = (tid & 1) * 16;
    int base_q = b * 128 + q;
    const float* Qrow = QKV + (size_t)base_q * 768;
    float sc = mk * l_mq[q] * 0.17677669529663687f;
    float w1q[16], w2q[16], ddq[8];
#pragma unroll
    for (int i = 0; i < 16; ++i) {
      w1q[i] = W1n[((size_t)base_q * 4 + 0) * 16 + i];
      w2q[i] = W2[((size_t)base_q * 4 + 0) * 16 + i];
    }
#pragma unroll
    for (int i = 0; i < 8; ++i) ddq[i] = DD[((size_t)base_q * 4 + 0) * 8 + i];
    float qh0[16] = {}, qh1[16] = {}, kh0[16] = {}, kh1[16] = {};
    for (int h = 0; h < 8; ++h) {
      float a0 = w1q[h], a1 = w1q[8 + h], b0 = l_wk[h], b1 = l_wk[8 + h];
#pragma unroll
      for (int j = 0; j < 16; ++j) {
        int c = h * 32 + f0 + j;
        float Y = Qrow[c] * l_kr[c] * sc;
        qh0[j] += Y * a0; qh1[j] += Y * a1;
        kh0[j] += Y * b0; kh1[j] += Y * b1;
      }
    }
    for (int h = 0; h < 8; ++h) {
      float dsum = 1.f + ddq[h] + l_wk[32 + h];
      float c0 = w2q[h], c1 = w2q[8 + h], d0 = l_wk[16 + h], d1 = l_wk[24 + h];
#pragma unroll
      for (int j = 0; j < 16; ++j) {
        int c = h * 32 + f0 + j;
        float Y = Qrow[c] * l_kr[c] * sc;
        l_log[q * 256 + c] = Y * dsum + qh0[j] * c0 + qh1[j] * c1 + kh0[j] * d0 + kh1[j] * d1;
      }
    }
  }
  __syncthreads();

  // Phase 2: outE = E + ((gate*logits) @ e_out_w + b) * mask   (WMMA, K=256)
  {
    int lane = tid & 31, wid = tid >> 5;
    for (int tile = wid; tile < 32; tile += 8) {
      int qt = tile >> 2, nt = tile & 3;
      f32x8 acc = {0.f, 0.f, 0.f, 0.f, 0.f, 0.f, 0.f, 0.f};
      for (int kt = 0; kt < 8; ++kt) {
        int r = qt * 16 + (lane & 15), hh = lane >> 4;
        bf16x16 af;
#pragma unroll
        for (int j = 0; j < 16; ++j) {
          int kk = kt * 32 + a_kmap(j, hh);
          af[j] = (bf16)(l_log[r * 256 + kk] * l_ge[kk]);
        }
        bf16x16 bfr = gather_b_bf16(eoutw + kt * 32 * 64 + nt * 16, 64, lane);
        acc = wmma_bf16(af, bfr, acc);
      }
      int col = nt * 16 + (lane & 15);
      float bv = eoutb[col];
      int rb = qt * 16 + ((lane >> 4) << 3);
#pragma unroll
      for (int j = 0; j < 8; ++j) {
        int q = rb + j;
        float m = mk * l_mq[q];
        size_t idx = ((size_t)base_k * 128 + q) * 64 + col;
        outE[idx] = Ein[idx] + (acc[j] + bv) * m;
      }
    }
  }
  __syncthreads();

  // Phase 3: softmax over q (masked), in place; thread owns one channel
  {
    int c = tid;
    float mmax = -3.4e38f;
    for (int q = 0; q < 128; ++q)
      if (l_mq[q] > 0.f) mmax = fmaxf(mmax, l_log[q * 256 + c]);
    float s = 0.f;
    for (int q = 0; q < 128; ++q) {
      float e = (l_mq[q] > 0.f) ? __expf(l_log[q * 256 + c] - mmax) : 0.f;
      l_log[q * 256 + c] = e;
      s += e;
    }
    float inv = (s > 0.f) ? 1.f / s : 0.f;
    for (int q = 0; q < 128; ++q) l_log[q * 256 + c] *= inv;
  }
  __syncthreads();

  // Phase 4: post cross_head_proj on probs, weighted-V accumulation
  {
    int q = tid >> 1, f0 = (tid & 1) * 16;
    int base_q = b * 128 + q;
    const float* Vrow = QKV + (size_t)base_q * 768 + 512;
    float w1q[16], w2q[16], ddq[8];
#pragma unroll
    for (int i = 0; i < 16; ++i) {
      w1q[i] = W1n[((size_t)base_q * 4 + 2) * 16 + i];
      w2q[i] = W2[((size_t)base_q * 4 + 2) * 16 + i];
    }
#pragma unroll
    for (int i = 0; i < 8; ++i) ddq[i] = DD[((size_t)base_q * 4 + 2) * 8 + i];
    float qh0[16] = {}, qh1[16] = {}, kh0[16] = {}, kh1[16] = {};
    for (int h = 0; h < 8; ++h) {
      float a0 = w1q[h], a1 = w1q[8 + h], b0 = l_wk[40 + h], b1 = l_wk[48 + h];
#pragma unroll
      for (int j = 0; j < 16; ++j) {
        float p = l_log[q * 256 + h * 32 + f0 + j];
        qh0[j] += p * a0; qh1[j] += p * a1;
        kh0[j] += p * b0; kh1[j] += p * b1;
      }
    }
    for (int h = 0; h < 8; ++h) {
      float dsum = 1.f + ddq[h] + l_wk[72 + h];
      float c0 = w2q[h], c1 = w2q[8 + h], d0 = l_wk[56 + h], d1 = l_wk[64 + h];
#pragma unroll
      for (int j = 0; j < 16; ++j) {
        int c = h * 32 + f0 + j;
        float p = l_log[q * 256 + c];
        float o = p * dsum + qh0[j] * c0 + qh1[j] * c1 + kh0[j] * d0 + kh1[j] * d1;
        atomicAdd(&l_wv[c], o * Vrow[c]);
      }
    }
  }
  __syncthreads();

  // Phase 5: dyn_bias = mean(wV); newX_pre = (gateN + dyn)*wV
  {
    l_red[tid] = l_wv[tid];
    __syncthreads();
    for (int s = 128; s > 0; s >>= 1) {
      if (tid < s) l_red[tid] += l_red[tid + s];
      __syncthreads();
    }
    float dyn = l_red[0] * (1.f / 256.f);
    newXpre[(size_t)base_k * 256 + tid] = (gateN[b * 256 + tid] + dyn) * l_wv[tid];
  }
}

// ---------------- row LayerNorm (in-place, optional residual) ----------------
__global__ void ln_kernel(float* __restrict__ io, const float* __restrict__ res,
                          const float* __restrict__ g, const float* __restrict__ b, float eps) {
  extern __shared__ char smraw[];
  float* red = (float*)smraw;
  int D = blockDim.x, c = threadIdx.x;
  size_t row = blockIdx.x;
  float v = io[row * D + c];
  if (res) v += res[row * D + c];
  red[c] = v;
  __syncthreads();
  for (int s = D >> 1; s > 0; s >>= 1) { if (c < s) red[c] += red[c + s]; __syncthreads(); }
  float mu = red[0] / (float)D;
  __syncthreads();
  float d = v - mu;
  red[c] = d * d;
  __syncthreads();
  for (int s = D >> 1; s > 0; s >>= 1) { if (c < s) red[c] += red[c + s]; __syncthreads(); }
  float var = red[0] / (float)D;
  io[row * D + c] = d * rsqrtf(var + eps) * g[c] + b[c];
}

// ---------------- fused FFN + residual + LN (in-place on io rows) ----------------
__global__ void __launch_bounds__(256) ff_ln_kernel(float* __restrict__ io,
    const bf16* __restrict__ W1, const float* __restrict__ b1,
    const bf16* __restrict__ W2, const float* __restrict__ b2,
    const float* __restrict__ g, const float* __restrict__ bln,
    int D, int H, float eps) {
  extern __shared__ char smraw[];
  bf16* smIn = (bf16*)smraw;
  bf16* smH = smIn + 16 * D;
  float* smOut = (float*)(smH + 16 * H);
  float* smRed = smOut + 16 * D;
  float* smMu = smRed + 256;
  float* smVar = smMu + 16;
  int tid = threadIdx.x, lane = tid & 31, wid = tid >> 5;
  size_t row0 = (size_t)blockIdx.x * 16;
  for (int i = tid; i < 16 * D; i += 256) {
    int r = i / D, c = i % D;
    smIn[i] = (bf16)io[(row0 + r) * D + c];
  }
  __syncthreads();
  int nt1 = H >> 4, kt1 = D >> 5;
  for (int t = wid; t < nt1; t += 8) {
    f32x8 acc = {0.f, 0.f, 0.f, 0.f, 0.f, 0.f, 0.f, 0.f};
    for (int kt = 0; kt < kt1; ++kt) {
      __builtin_prefetch(W1 + (size_t)kt * 32 * H + t * 16, 0, 1);
      bf16x16 af = gather_a_bf16(smIn + kt * 32, D, lane);
      bf16x16 bfr = gather_b_bf16(W1 + (size_t)kt * 32 * H + t * 16, H, lane);
      acc = wmma_bf16(af, bfr, acc);
    }
    int col = t * 16 + (lane & 15);
    int rb = (lane >> 4) << 3;
    float bv = b1[col];
#pragma unroll
    for (int j = 0; j < 8; ++j) smH[(rb + j) * H + col] = (bf16)siluf_(acc[j] + bv);
  }
  __syncthreads();
  int nt2 = D >> 4, kt2 = H >> 5;
  for (int t = wid; t < nt2; t += 8) {
    f32x8 acc = {0.f, 0.f, 0.f, 0.f, 0.f, 0.f, 0.f, 0.f};
    for (int kt = 0; kt < kt2; ++kt) {
      bf16x16 af = gather_a_bf16(smH + kt * 32, H, lane);
      bf16x16 bfr = gather_b_bf16(W2 + (size_t)kt * 32 * D + t * 16, D, lane);
      acc = wmma_bf16(af, bfr, acc);
    }
    int col = t * 16 + (lane & 15);
    int rb = (lane >> 4) << 3;
    float bv = b2[col];
#pragma unroll
    for (int j = 0; j < 8; ++j)
      smOut[(rb + j) * D + col] = acc[j] + bv + io[(row0 + rb + j) * D + col];
  }
  __syncthreads();
  // per-row LN (16 rows x 16 lanes each)
  int r16 = tid >> 4, s = tid & 15, cpt = D >> 4;
  float ps = 0.f;
  for (int t = 0; t < cpt; ++t) ps += smOut[r16 * D + s * cpt + t];
  smRed[r16 * 16 + s] = ps;
  __syncthreads();
  if (s == 0) { float m = 0.f; for (int u = 0; u < 16; ++u) m += smRed[r16 * 16 + u]; smMu[r16] = m / (float)D; }
  __syncthreads();
  float mu = smMu[r16];
  ps = 0.f;
  for (int t = 0; t < cpt; ++t) { float d = smOut[r16 * D + s * cpt + t] - mu; ps += d * d; }
  smRed[r16 * 16 + s] = ps;
  __syncthreads();
  if (s == 0) { float m = 0.f; for (int u = 0; u < 16; ++u) m += smRed[r16 * 16 + u]; smVar[r16] = m / (float)D; }
  __syncthreads();
  float inv = rsqrtf(smVar[r16] + eps);
  for (int t = 0; t < cpt; ++t) {
    int c = s * cpt + t;
    io[(row0 + r16) * D + c] = (smOut[r16 * D + c] - mu) * inv * g[c] + bln[c];
  }
}

// ---------------- E symmetrize in place ----------------
__global__ void sym_kernel(float* __restrict__ Eo) {
  int i = blockIdx.x, j = blockIdx.y, b = blockIdx.z, c = threadIdx.x;
  if (j < i) return;
  size_t i1 = (((size_t)b * 128 + i) * 128 + j) * 64 + c;
  size_t i2 = (((size_t)b * 128 + j) * 128 + i) * 64 + c;
  float a = Eo[i1], d = Eo[i2];
  float m = 0.5f * (a + d);
  Eo[i1] = m;
  if (j != i) Eo[i2] = m;
}

// ---------------- X / E statistics for y path ----------------
__global__ void xstats_kernel(const float* __restrict__ X, float* __restrict__ xz) {
  int b = blockIdx.x, c = threadIdx.x;
  float s = 0.f, s2 = 0.f, mn = 3.4e38f, mx = -3.4e38f;
  for (int nidx = 0; nidx < 128; ++nidx) {
    float v = X[((size_t)b * 128 + nidx) * 256 + c];
    s += v; s2 += v * v; mn = fminf(mn, v); mx = fmaxf(mx, v);
  }
  float mu = s / 128.f;
  xz[b * 1024 + c] = mu;
  xz[b * 1024 + 256 + c] = mn;
  xz[b * 1024 + 512 + c] = mx;
  xz[b * 1024 + 768 + c] = sqrtf(fmaxf((s2 - 128.f * mu * mu) / 127.f, 0.f));
}
__global__ void estats_kernel(const float* __restrict__ E, float* __restrict__ ez) {
  __shared__ float rs[256], rs2[256], rmn[256], rmx[256];
  int c = blockIdx.x, b = blockIdx.y, t = threadIdx.x;
  float s = 0.f, s2 = 0.f, mn = 3.4e38f, mx = -3.4e38f;
  for (int i = t; i < 16384; i += 256) {
    float v = E[(((size_t)b * 16384) + i) * 64 + c];
    s += v; s2 += v * v; mn = fminf(mn, v); mx = fmaxf(mx, v);
  }
  rs[t] = s; rs2[t] = s2; rmn[t] = mn; rmx[t] = mx;
  __syncthreads();
  for (int st = 128; st > 0; st >>= 1) {
    if (t < st) {
      rs[t] += rs[t + st]; rs2[t] += rs2[t + st];
      rmn[t] = fminf(rmn[t], rmn[t + st]); rmx[t] = fmaxf(rmx[t], rmx[t + st]);
    }
    __syncthreads();
  }
  if (t == 0) {
    float mu = rs[0] / 16384.f;
    ez[b * 256 + c] = mu;
    ez[b * 256 + 64 + c] = rmn[0];
    ez[b * 256 + 128 + c] = rmx[0];
    ez[b * 256 + 192 + c] = sqrtf(fmaxf((rs2[0] - 16384.f * mu * mu) / 16383.f, 0.f));
  }
}

// ---------------- y head (tiny; one block per b) ----------------
__global__ void yhead_kernel(const float* __restrict__ y, const float* __restrict__ xz,
                             const float* __restrict__ ez,
                             const float* yyw, const float* yyb,
                             const float* xyw, const float* xyb,
                             const float* eyw, const float* eyb,
                             const float* w1, const float* bb1,
                             const float* w2, const float* bb2,
                             const float* g1, const float* be1,
                             const float* ly1w, const float* ly1b,
                             const float* ly2w, const float* ly2b,
                             const float* g2, const float* be2,
                             float* __restrict__ outy) {
  __shared__ float sny[128], st1[128], syr[128], red[128];
  __shared__ float sh[2048];
  int b = blockIdx.x, d = threadIdx.x;
  float acc = yyb[d] + xyb[d] + eyb[d];
  for (int j = 0; j < 128; ++j) acc += y[b * 128 + j] * yyw[j * 128 + d];
  for (int j = 0; j < 1024; ++j) acc += xz[b * 1024 + j] * xyw[j * 128 + d];
  for (int j = 0; j < 256; ++j) acc += ez[b * 256 + j] * eyw[j * 128 + d];
  sny[d] = acc;
  __syncthreads();
  float t1 = bb1[d];
  for (int j = 0; j < 128; ++j) t1 += sny[j] * w1[j * 128 + d];
  st1[d] = fmaxf(t1, 0.f);
  __syncthreads();
  float ny = bb2[d];
  for (int j = 0; j < 128; ++j) ny += st1[j] * w2[j * 128 + d];
  float v = y[b * 128 + d] + ny;
  // LN1
  red[d] = v; __syncthreads();
  for (int st = 64; st > 0; st >>= 1) { if (d < st) red[d] += red[d + st]; __syncthreads(); }
  float mu = red[0] * (1.f / 128.f); __syncthreads();
  float dv = v - mu;
  red[d] = dv * dv; __syncthreads();
  for (int st = 64; st > 0; st >>= 1) { if (d < st) red[d] += red[d + st]; __syncthreads(); }
  float var = red[0] * (1.f / 128.f); __syncthreads();
  float yr = dv * rsqrtf(var + 1e-5f) * g1[d] + be1[d];
  syr[d] = yr;
  __syncthreads();
  for (int t = d; t < 2048; t += 128) {
    float hv = ly1b[t];
    for (int j = 0; j < 128; ++j) hv += syr[j] * ly1w[j * 2048 + t];
    sh[t] = siluf_(hv);
  }
  __syncthreads();
  float ff = ly2b[d];
  for (int j = 0; j < 2048; ++j) ff += sh[j] * ly2w[j * 128 + d];
  float v2 = yr + ff;
  // LN2
  red[d] = v2; __syncthreads();
  for (int st = 64; st > 0; st >>= 1) { if (d < st) red[d] += red[d + st]; __syncthreads(); }
  float mu2 = red[0] * (1.f / 128.f); __syncthreads();
  float dv2 = v2 - mu2;
  red[d] = dv2 * dv2; __syncthreads();
  for (int st = 64; st > 0; st >>= 1) { if (d < st) red[d] += red[d + st]; __syncthreads(); }
  float var2 = red[0] * (1.f / 128.f);
  outy[b * 128 + d] = dv2 * rsqrtf(var2 + 1e-5f) * g2[d] + be2[d];
}

// ---------------- host orchestration ----------------
extern "C" void kernel_launch(void* const* d_in, const int* in_sizes, int n_in,
                              void* d_out, int out_size, void* d_ws, size_t ws_size,
                              hipStream_t stream) {
  (void)in_sizes; (void)n_in; (void)out_size; (void)ws_size;
  const float* X = (const float*)d_in[0];
  const float* E = (const float*)d_in[1];
  const float* y = (const float*)d_in[2];
  const float* mask = (const float*)d_in[3];
  const float* wqkv_w = (const float*)d_in[4];
  const float* dw_m = (const float*)d_in[5];
  const float* qkw_m = (const float*)d_in[6];
  const float* node_gate_w = (const float*)d_in[7];
  const float* node_gate_b = (const float*)d_in[8];
  const float* edge_gate_w = (const float*)d_in[9];
  const float* edge_gate_b = (const float*)d_in[10];
  const float* y_y_w = (const float*)d_in[11];
  const float* y_y_b = (const float*)d_in[12];
  const float* x_y_w = (const float*)d_in[13];
  const float* x_y_b = (const float*)d_in[14];
  const float* e_y_w = (const float*)d_in[15];
  const float* e_y_b = (const float*)d_in[16];
  const float* x_out_w = (const float*)d_in[17];
  const float* x_out_b = (const float*)d_in[18];
  const float* e_out_w = (const float*)d_in[19];
  const float* e_out_b = (const float*)d_in[20];
  const float* y_out_w1 = (const float*)d_in[21];
  const float* y_out_b1 = (const float*)d_in[22];
  const float* y_out_w2 = (const float*)d_in[23];
  const float* y_out_b2 = (const float*)d_in[24];
  const float* linX1_w = (const float*)d_in[25];
  const float* linX1_b = (const float*)d_in[26];
  const float* linX2_w = (const float*)d_in[27];
  const float* linX2_b = (const float*)d_in[28];
  const float* normX1_g = (const float*)d_in[29];
  const float* normX1_b = (const float*)d_in[30];
  const float* normX2_g = (const float*)d_in[31];
  const float* normX2_b = (const float*)d_in[32];
  const float* linE1_w = (const float*)d_in[33];
  const float* linE1_b = (const float*)d_in[34];
  const float* linE2_w = (const float*)d_in[35];
  const float* linE2_b = (const float*)d_in[36];
  const float* normE1_g = (const float*)d_in[37];
  const float* normE1_b = (const float*)d_in[38];
  const float* normE2_g = (const float*)d_in[39];
  const float* normE2_b = (const float*)d_in[40];
  const float* lin_y1_w = (const float*)d_in[41];
  const float* lin_y1_b = (const float*)d_in[42];
  const float* lin_y2_w = (const float*)d_in[43];
  const float* lin_y2_b = (const float*)d_in[44];
  const float* norm_y1_g = (const float*)d_in[45];
  const float* norm_y1_b = (const float*)d_in[46];
  const float* norm_y2_g = (const float*)d_in[47];
  const float* norm_y2_b = (const float*)d_in[48];

  float* outX = (float*)d_out;
  float* outE = outX + (size_t)8 * 128 * 256;
  float* outy = outE + (size_t)8 * 128 * 128 * 64;

  char* wp = (char*)d_ws;
  auto alloc_bf = [&](size_t n) { bf16* r = (bf16*)wp; wp += ((n * 2 + 255) / 256) * 256; return r; };
  auto alloc_f = [&](size_t n) { float* r = (float*)wp; wp += ((n * 4 + 255) / 256) * 256; return r; };

  bf16* wqkv_bf = alloc_bf(256 * 768);
  bf16* dwm_bf = alloc_bf(256 * 160);
  bf16* eout_bf = alloc_bf(256 * 64);
  bf16* xout_bf = alloc_bf(256 * 256);
  bf16* lx1_bf = alloc_bf(256 * 2048);
  bf16* lx2_bf = alloc_bf(2048 * 256);
  bf16* le1_bf = alloc_bf(64 * 128);
  bf16* le2_bf = alloc_bf(128 * 64);
  float* QKVf = alloc_f((size_t)1024 * 768);
  float* dwraw = alloc_f((size_t)1024 * 160);
  float* W1n = alloc_f((size_t)1024 * 64);
  float* W2v = alloc_f((size_t)1024 * 64);
  float* DDv = alloc_f((size_t)1024 * 32);
  float* gateN = alloc_f(8 * 256);
  float* gateE = alloc_f(8 * 256);
  float* newXpre = alloc_f((size_t)1024 * 256);
  float* xz = alloc_f(8 * 1024);
  float* ez = alloc_f(8 * 256);

  auto cvt = [&](const float* s, bf16* d, int n) {
    cvt_kernel<<<(n + 255) / 256, 256, 0, stream>>>(s, d, n);
  };
  cvt(wqkv_w, wqkv_bf, 256 * 768);
  cvt(dw_m, dwm_bf, 256 * 160);
  cvt(e_out_w, eout_bf, 256 * 64);
  cvt(x_out_w, xout_bf, 256 * 256);
  cvt(linX1_w, lx1_bf, 256 * 2048);
  cvt(linX2_w, lx2_bf, 2048 * 256);
  cvt(linE1_w, le1_bf, 64 * 128);
  cvt(linE2_w, le2_bf, 128 * 64);

  // QKV = (X @ wqkv) * node_mask rows
  gemm_kernel<<<dim3(768 / 16, 1024 / 16), 32, 0, stream>>>(X, 256, wqkv_bf, 768, nullptr, mask, QKVf, 768, 256, 0);
  // dw_raw = X @ dw_m
  gemm_kernel<<<dim3(160 / 16, 1024 / 16), 32, 0, stream>>>(X, 256, dwm_bf, 160, nullptr, nullptr, dwraw, 160, 256, 0);
  dw_kernel<<<1024, 128, 0, stream>>>(dwraw, qkw_m, W1n, W2v, DDv);
  gate_kernel<<<8, 256, 0, stream>>>(y, node_gate_w, node_gate_b, edge_gate_w, edge_gate_b, gateN, gateE);

  size_t msm = (size_t)(128 * 256 + 256 + 128 + 256 + 256 + 256 + 96) * sizeof(float);
  edge_attn_kernel<<<1024, 256, msm, stream>>>(QKVf, W1n, W2v, DDv, gateN, gateE,
                                               eout_bf, e_out_b, E, mask, outE, newXpre);

  // newX = (newXpre @ x_out_w + b) * mask ; then Xr = LN(X + newX) in place; then fused FFN+LN
  gemm_kernel<<<dim3(256 / 16, 1024 / 16), 32, 0, stream>>>(newXpre, 256, xout_bf, 256, x_out_b, mask, outX, 256, 256, 0);
  ln_kernel<<<1024, 256, 256 * sizeof(float), stream>>>(outX, X, normX1_g, normX1_b, 1e-5f);
  size_t ffx = (size_t)16 * 256 * 2 + (size_t)16 * 2048 * 2 + (size_t)16 * 256 * 4 + (256 + 32) * 4;
  ff_ln_kernel<<<64, 256, ffx, stream>>>(outX, lx1_bf, linX1_b, lx2_bf, linX2_b, normX2_g, normX2_b, 256, 2048, 1e-5f);

  // E path: outE already holds E + newE; LN, fused FFN+LN, symmetrize (all in place)
  ln_kernel<<<131072, 64, 64 * sizeof(float), stream>>>(outE, nullptr, normE1_g, normE1_b, 1e-5f);
  size_t ffe = (size_t)16 * 64 * 2 + (size_t)16 * 128 * 2 + (size_t)16 * 64 * 4 + (256 + 32) * 4;
  ff_ln_kernel<<<8192, 256, ffe, stream>>>(outE, le1_bf, linE1_b, le2_bf, linE2_b, normE2_g, normE2_b, 64, 128, 1e-5f);
  sym_kernel<<<dim3(128, 128, 8), 64, 0, stream>>>(outE);

  // y path
  xstats_kernel<<<8, 256, 0, stream>>>(X, xz);
  estats_kernel<<<dim3(64, 8), 256, 0, stream>>>(E, ez);
  yhead_kernel<<<8, 128, 0, stream>>>(y, xz, ez, y_y_w, y_y_b, x_y_w, x_y_b, e_y_w, e_y_b,
                                      y_out_w1, y_out_b1, y_out_w2, y_out_b2,
                                      norm_y1_g, norm_y1_b, lin_y1_w, lin_y1_b,
                                      lin_y2_w, lin_y2_b, norm_y2_g, norm_y2_b, outy);
}
